// MultiHeadAttention_52604759441868
// MI455X (gfx1250) — matrix operand order repaired and checked by reading, not compiled, not run on hardware
//
#include <hip/hip_runtime.h>
#include <cstdint>
#include <cstddef>

#define DEV __device__ __forceinline__

typedef __attribute__((ext_vector_type(16))) __bf16 v16bf;
typedef __attribute__((ext_vector_type(8)))  float  v8f;

// ---------------- problem constants ----------------
static constexpr int BB  = 4;
static constexpr int LL  = 2048;
static constexpr int DM  = 1024;   // model dim
static constexpr int HH  = 16;     // heads
static constexpr int DD  = 64;     // head dim
static constexpr int RR  = 256;    // random features
static constexpr int CC  = 64;     // chunk length
static constexpr int NC  = LL / CC;          // 32 chunks per (b,h)
static constexpr int MROWS = BB * LL;        // 8192
static constexpr float ORF_SCALE = 0.08838834764831845f; // sqrt(2/256)

// ---------------- bf16 helpers (let clang pick native cvt, RNE) -------------
DEV uint16_t f2bf(float f) {
  __bf16 h = (__bf16)f;
  return __builtin_bit_cast(uint16_t, h);
}
DEV float bf2f(uint16_t h) { return __uint_as_float(((uint32_t)h) << 16); }
DEV uint32_t packbf2(float lo, float hi) {
  return (uint32_t)f2bf(lo) | ((uint32_t)f2bf(hi) << 16);
}
// paired loaders for LDS staging (u32 = two bf16)
DEV uint32_t ld_pair(const float* p)    { return packbf2(p[0], p[1]); }
DEV uint32_t ld_pair(const uint16_t* p) { return *(const uint32_t*)p; }
DEV void st_out(float* o,   size_t i, float v) { o[i] = v; }
DEV void st_out(uint16_t* o, size_t i, float v) { o[i] = f2bf(v); }

union FragU { uint32_t u[8]; v16bf v; };

DEV v8f vzero() {
  v8f z;
#pragma unroll
  for (int i = 0; i < 8; ++i) z[i] = 0.0f;
  return z;
}

DEV v8f wmma_bf(v16bf a, v16bf b, v8f c) {
  return __builtin_amdgcn_wmma_f32_16x16x32_bf16(false, a, false, b, (short)0, c,
                                                 false, false);
}

// A-frag (16x32 bf16) from row-major [m][k] array (ld in u16 units, even pad).
// Layout: lanes 0-15 -> K 0..15 ; lanes 16-31 -> K 16..31.
DEV v16bf ldA(const uint16_t* s, int ld, int mB, int kB, int lane) {
  FragU f;
  const uint16_t* row = s + (size_t)(mB + (lane & 15)) * ld;
  int kh = (lane >> 4) * 16;
#pragma unroll
  for (int i = 0; i < 8; ++i) {
    int k = kB + 2 * i + kh;
    f.u[i] = *(const uint32_t*)(row + k);
  }
  return f.v;
}
// B-frag (32x16 bf16) from [n][k] array: lane n = lane&15, K 0..15 / 16..31.
DEV v16bf ldB(const uint16_t* s, int ld, int nB, int kB, int lane) {
  FragU f;
  const uint16_t* row = s + (size_t)(nB + (lane & 15)) * ld;
  int kh = (lane >> 4) * 16;
#pragma unroll
  for (int v = 0; v < 8; ++v) {
    int k = kB + 2 * v + kh;
    f.u[v] = *(const uint32_t*)(row + k);
  }
  return f.v;
}
// B-frag from global bf16 [n][k] (row stride ld u16, 4B-aligned pairs).
DEV v16bf gB_bf(const uint16_t* g, int ld, int nB, int kB, int lane) {
  FragU f;
  const uint16_t* row = g + (size_t)(nB + (lane & 15)) * ld;
  int kh = (lane >> 4) * 16;
#pragma unroll
  for (int v = 0; v < 8; ++v) {
    int k = kB + 2 * v + kh;
    f.u[v] = *(const uint32_t*)(row + k);
  }
  return f.v;
}

// ================= prep: omega f32 -> bf16 (once) =================
__global__ __launch_bounds__(256) void cvt_omega_kernel(
    const float* __restrict__ src, uint16_t* __restrict__ dst) {
  int i = blockIdx.x * 256 + threadIdx.x;  // RR*DD = 16384 elements
  dst[i] = f2bf(src[i]);
}

// ================= GEMM: O[M,N] = X[M,K] @ W[N,K]^T =================
// 128x128 block tile, 8 waves, K staged 32 wide through LDS (u32 pairs).
template <typename XT, typename OT>
__global__ __launch_bounds__(256) void gemm_xwt_kernel(
    const XT* __restrict__ X, const float* __restrict__ W, OT* __restrict__ O,
    int M, int K, int N) {
  __shared__ uint16_t As[128][36];  // 36*2B row = 18 words: conflict-free
  __shared__ uint16_t Bs[128][36];
  const int tid = threadIdx.x, lane = tid & 31, wave = tid >> 5;
  const int m0 = blockIdx.x * 128, n0 = blockIdx.y * 128;

  v8f acc[8];
#pragma unroll
  for (int i = 0; i < 8; ++i) acc[i] = vzero();

  for (int kk = 0; kk < K; kk += 32) {
#pragma unroll
    for (int i = 0; i < 8; ++i) {  // 2048 u32 pairs per buffer, 8 per thread
      int e = i * 256 + tid;
      int r = e >> 4, c2 = (e & 15) * 2;
      *(uint32_t*)&As[r][c2] = ld_pair(&X[(size_t)(m0 + r) * K + kk + c2]);
      *(uint32_t*)&Bs[r][c2] = ld_pair(&W[(size_t)(n0 + r) * K + kk + c2]);
    }
    __syncthreads();
    v16bf af = ldA(&As[0][0], 36, wave * 16, 0, lane);
#pragma unroll
    for (int nt = 0; nt < 8; ++nt) {
      v16bf bf = ldB(&Bs[0][0], 36, nt * 16, 0, lane);
      acc[nt] = wmma_bf(af, bf, acc[nt]);
    }
    __syncthreads();
  }
  const int mb = m0 + wave * 16 + ((lane >> 4) * 8);
#pragma unroll
  for (int nt = 0; nt < 8; ++nt) {
    int n = n0 + nt * 16 + (lane & 15);
#pragma unroll
    for (int r = 0; r < 8; ++r)
      st_out(O, (size_t)(mb + r) * N + n, acc[nt][r]);
  }
}

// ================= Phase A: per-chunk k' = orf(k), Mc = k'^T V, zc ==========
__global__ __launch_bounds__(256) void phase_a_kernel(
    const uint16_t* __restrict__ kbf, const uint16_t* __restrict__ vbf,
    const uint16_t* __restrict__ omg, const float* __restrict__ bvec,
    uint16_t* __restrict__ kp_ws, uint16_t* __restrict__ mc_ws,
    float* __restrict__ zc_ws) {
  __shared__ uint16_t Ks[64][68];    // [t][d]
  __shared__ uint16_t VsT[64][68];   // [d][t]
  __shared__ uint16_t kpT[256][68];  // [r][t]
  const int tid = threadIdx.x, lane = tid & 31, wave = tid >> 5;
  const int bhc = blockIdx.x;
  const int c = bhc & (NC - 1), h = (bhc >> 5) & (HH - 1), b = bhc >> 9;
  const int t0 = c * CC;

  // stage K chunk (u32 pairs) and V^T chunk (transposed u16 pair)
#pragma unroll
  for (int i = 0; i < 8; ++i) {
    int e = i * 256 + tid;
    int t = e >> 5, d2 = (e & 31) * 2;
    size_t g = ((size_t)(b * LL + t0 + t)) * DM + h * DD + d2;
    *(uint32_t*)&Ks[t][d2] = *(const uint32_t*)&kbf[g];
    uint32_t vv = *(const uint32_t*)&vbf[g];
    VsT[d2][t] = (uint16_t)(vv & 0xffffu);
    VsT[d2 + 1][t] = (uint16_t)(vv >> 16);
  }
  __syncthreads();

  // k' = ORF(K @ omega^T + b): 4 m-tiles x 16 r-tiles, 8 per wave
  const size_t kpbase = (size_t)bhc * CC * RR;
#pragma unroll
  for (int i = 0; i < 8; ++i) {
    int tile = wave * 8 + i;
    int mt = tile & 3, rt = tile >> 2;
    v8f acc = vzero();
#pragma unroll
    for (int ks = 0; ks < 64; ks += 32) {
      v16bf af = ldA(&Ks[0][0], 68, mt * 16, ks, lane);
      v16bf bf = gB_bf(omg, DD, rt * 16, ks, lane);
      acc = wmma_bf(af, bf, acc);
    }
    int r = rt * 16 + (lane & 15);
    float bb = bvec[r];
#pragma unroll
    for (int reg = 0; reg < 8; ++reg) {
      int t = mt * 16 + reg + ((lane >> 4) * 8);
      uint16_t bits = f2bf(ORF_SCALE * __cosf(acc[reg] + bb));
      kpT[r][t] = bits;
      kp_ws[kpbase + (size_t)t * RR + r] = bits;  // [t][r] for phase C
    }
  }
  __syncthreads();

  // zc[r] = sum_t k'[t,r]
  {
    float s = 0.0f;
#pragma unroll 8
    for (int t = 0; t < CC; ++t) s += bf2f(kpT[tid][t]);
    zc_ws[(size_t)bhc * RR + tid] = s;
  }

  // Mc = k'^T @ V : 16 r-tiles x 4 d-tiles, 8 per wave
#pragma unroll
  for (int i = 0; i < 8; ++i) {
    int tile = wave * 8 + i;
    int rt = tile >> 2, dt = tile & 3;
    v8f acc = vzero();
#pragma unroll
    for (int ks = 0; ks < 64; ks += 32) {
      v16bf af = ldA(&kpT[0][0], 68, rt * 16, ks, lane);
      v16bf bf = ldB(&VsT[0][0], 68, dt * 16, ks, lane);
      acc = wmma_bf(af, bf, acc);
    }
#pragma unroll
    for (int reg = 0; reg < 8; ++reg) {
      int r = rt * 16 + reg + ((lane >> 4) * 8);
      int d = dt * 16 + (lane & 15);
      mc_ws[((size_t)bhc * RR + r) * DD + d] = f2bf(acc[reg]);
    }
  }
}

// ================= Phase B: exclusive prefix scan over chunks ==============
__global__ __launch_bounds__(256) void scan_kernel(
    const uint16_t* __restrict__ mc, const float* __restrict__ zc,
    uint16_t* __restrict__ spt, float* __restrict__ zprev) {
  const int bh = blockIdx.x, tid = threadIdx.x;
  float S[64];
#pragma unroll
  for (int i = 0; i < 64; ++i) S[i] = 0.0f;
  float zr = 0.0f;
  for (int c = 0; c < NC; ++c) {
    size_t cb = (size_t)bh * NC + c;
    zprev[cb * RR + tid] = zr;
    zr += zc[cb * RR + tid];
#pragma unroll
    for (int i = 0; i < 64; ++i) {
      int p = i * 256 + tid;
      int r = p >> 6, d = p & 63;
      spt[(cb * DD + d) * RR + r] = f2bf(S[i]);
      S[i] += bf2f(mc[cb * (RR * DD) + p]);
    }
  }
}

// ================= Phase C: per-chunk output =================
__global__ __launch_bounds__(256) void phase_c_kernel(
    const uint16_t* __restrict__ qbf, const uint16_t* __restrict__ vbf,
    const uint16_t* __restrict__ omg, const float* __restrict__ bvec,
    const uint16_t* __restrict__ kp_ws, const uint16_t* __restrict__ spt,
    const float* __restrict__ zprev, uint16_t* __restrict__ attn) {
  __shared__ uint16_t Xs[64][68];     // q chunk [t][d]
  __shared__ uint16_t VsT[64][68];    // [d][t]
  __shared__ uint16_t qps[64][260];   // q' [t][r]
  __shared__ uint16_t as_[64][68];    // masked A [t][t']
  __shared__ float zl[256];
  __shared__ float denp[256];
  __shared__ float den[64];
  const int tid = threadIdx.x, lane = tid & 31, wave = tid >> 5;
  const int bhc = blockIdx.x;
  const int c = bhc & (NC - 1), h = (bhc >> 5) & (HH - 1), b = bhc >> 9;
  const int t0 = c * CC;
  const size_t kpbase = (size_t)bhc * CC * RR;
  const size_t spbase = (size_t)bhc * DD * RR;

#pragma unroll
  for (int i = 0; i < 8; ++i) {
    int e = i * 256 + tid;
    int t = e >> 5, d2 = (e & 31) * 2;
    size_t g = ((size_t)(b * LL + t0 + t)) * DM + h * DD + d2;
    *(uint32_t*)&Xs[t][d2] = *(const uint32_t*)&qbf[g];
    uint32_t vv = *(const uint32_t*)&vbf[g];
    VsT[d2][t] = (uint16_t)(vv & 0xffffu);
    VsT[d2 + 1][t] = (uint16_t)(vv >> 16);
  }
  zl[tid] = zprev[(size_t)bhc * RR + tid];
  __syncthreads();

  // q' = ORF(Q @ omega^T + b)
#pragma unroll
  for (int i = 0; i < 8; ++i) {
    int tile = wave * 8 + i;
    int mt = tile & 3, rt = tile >> 2;
    v8f acc = vzero();
#pragma unroll
    for (int ks = 0; ks < 64; ks += 32) {
      v16bf af = ldA(&Xs[0][0], 68, mt * 16, ks, lane);
      v16bf bf = gB_bf(omg, DD, rt * 16, ks, lane);
      acc = wmma_bf(af, bf, acc);
    }
    int r = rt * 16 + (lane & 15);
    float bb = bvec[r];
#pragma unroll
    for (int reg = 0; reg < 8; ++reg) {
      int t = mt * 16 + reg + ((lane >> 4) * 8);
      qps[t][r] = f2bf(ORF_SCALE * __cosf(acc[reg] + bb));
    }
  }
  __syncthreads();

  // A = tril(q' k'^T): 4x4 tiles, 2 per wave; k' B-frags straight from global
#pragma unroll
  for (int i = 0; i < 2; ++i) {
    int tile = wave * 2 + i;
    int mt = tile >> 2, nt = tile & 3;
    v8f acc = vzero();
#pragma unroll
    for (int rs = 0; rs < 8; ++rs) {
      v16bf af = ldA(&qps[0][0], 260, mt * 16, rs * 32, lane);
      v16bf bf = gB_bf(kp_ws + kpbase, RR, nt * 16, rs * 32, lane);
      acc = wmma_bf(af, bf, acc);
    }
    int tp = nt * 16 + (lane & 15);
#pragma unroll
    for (int reg = 0; reg < 8; ++reg) {
      int t = mt * 16 + reg + ((lane >> 4) * 8);
      as_[t][tp] = f2bf((tp <= t) ? acc[reg] : 0.0f);  // inclusive causal mask
    }
  }
  __syncthreads();

  // den[t] = max(q'.z_prev + rowsum(A), 1e-6) + 1e-6   (A already masked)
  {
    int t = tid >> 2, part = tid & 3;
    float s = 0.0f;
    for (int r = part * 64; r < part * 64 + 64; ++r) s += bf2f(qps[t][r]) * zl[r];
    for (int tp = part * 16; tp < part * 16 + 16; ++tp) s += bf2f(as_[t][tp]);
    denp[tid] = s;
  }
  __syncthreads();
  if (tid < 64) {
    float s = denp[4 * tid] + denp[4 * tid + 1] + denp[4 * tid + 2] + denp[4 * tid + 3];
    den[tid] = fmaxf(s, 1e-6f) + 1e-6f;
  }
  __syncthreads();

  // num = q' @ S_prev + A @ V ; out = num / den
#pragma unroll
  for (int i = 0; i < 2; ++i) {
    int tile = wave * 2 + i;
    int mt = tile >> 2, dt = tile & 3;
    v8f acc = vzero();
#pragma unroll
    for (int rs = 0; rs < 8; ++rs) {
      v16bf af = ldA(&qps[0][0], 260, mt * 16, rs * 32, lane);
      v16bf bf = gB_bf(spt + spbase, RR, dt * 16, rs * 32, lane);
      acc = wmma_bf(af, bf, acc);
    }
#pragma unroll
    for (int ks = 0; ks < 64; ks += 32) {
      v16bf af = ldA(&as_[0][0], 68, mt * 16, ks, lane);
      v16bf bf = ldB(&VsT[0][0], 68, dt * 16, ks, lane);
      acc = wmma_bf(af, bf, acc);
    }
    int d = dt * 16 + (lane & 15);
#pragma unroll
    for (int reg = 0; reg < 8; ++reg) {
      int t = mt * 16 + reg + ((lane >> 4) * 8);
      float o = acc[reg] / den[t];
      attn[((size_t)(b * LL + t0 + t)) * DM + h * DD + d] = f2bf(o);
    }
  }
}

// ================= residual + LayerNorm =================
__global__ __launch_bounds__(256) void ln_kernel(
    const float* __restrict__ pre, const float* __restrict__ resid,
    const float* __restrict__ gamma, const float* __restrict__ beta,
    float* __restrict__ out) {
  __shared__ float rs[256], rss[256];
  const int row = blockIdx.x, tid = threadIdx.x;
  float x[4];
  float s = 0.0f, ss = 0.0f;
#pragma unroll
  for (int i = 0; i < 4; ++i) {
    int idx = tid + i * 256;
    float v = pre[(size_t)row * DM + idx] + resid[(size_t)row * DM + idx];
    x[i] = v;
    s += v;
    ss += v * v;
  }
  rs[tid] = s;
  rss[tid] = ss;
  __syncthreads();
  for (int st = 128; st > 0; st >>= 1) {
    if (tid < st) { rs[tid] += rs[tid + st]; rss[tid] += rss[tid + st]; }
    __syncthreads();
  }
  float mean = rs[0] * (1.0f / DM);
  float var = rss[0] * (1.0f / DM) - mean * mean;
  float inv = rsqrtf(var + 1e-5f);
#pragma unroll
  for (int i = 0; i < 4; ++i) {
    int idx = tid + i * 256;
    out[(size_t)row * DM + idx] = (x[i] - mean) * inv * gamma[idx] + beta[idx];
  }
}

// ================= host launch =================
extern "C" void kernel_launch(void* const* d_in, const int* in_sizes, int n_in,
                              void* d_out, int out_size, void* d_ws,
                              size_t ws_size, hipStream_t stream) {
  (void)in_sizes; (void)n_in; (void)out_size;
  const float* pre_q = (const float*)d_in[0];
  const float* pre_k = (const float*)d_in[1];
  const float* pre_v = (const float*)d_in[2];
  const float* wq    = (const float*)d_in[3];
  const float* wk    = (const float*)d_in[4];
  const float* wv    = (const float*)d_in[5];
  const float* wo    = (const float*)d_in[6];
  const float* gamma = (const float*)d_in[7];
  const float* beta  = (const float*)d_in[8];
  const float* omega = (const float*)d_in[9];
  const float* bvec  = (const float*)d_in[10];
  float* out = (float*)d_out;

  // workspace layout (bytes)
  uint8_t* ws = (uint8_t*)d_ws;
  const size_t SZ_QKV  = (size_t)MROWS * DM * 2;              // 16 MB each
  const size_t SZ_KP   = (size_t)BB * HH * NC * CC * RR * 2;  // 64 MB
  const size_t SZ_MC   = (size_t)BB * HH * NC * RR * DD * 2;  // 64 MB
  const size_t SZ_SPT  = SZ_MC;                               // 64 MB
  const size_t SZ_ZC   = (size_t)BB * HH * NC * RR * 4;       // 2 MB
  const size_t SZ_PRE  = (size_t)MROWS * DM * 4;              // 32 MB
  const size_t SZ_OMG  = (size_t)RR * DD * 2;                 // 32 KB
  size_t off = 0;
  auto take = [&](size_t n) { uint8_t* p = ws + off; off += (n + 255) & ~size_t(255); return p; };
  uint16_t* q_bf   = (uint16_t*)take(SZ_QKV);
  uint16_t* k_bf   = (uint16_t*)take(SZ_QKV);
  uint16_t* v_bf   = (uint16_t*)take(SZ_QKV);
  uint16_t* kp_ws  = (uint16_t*)take(SZ_KP);
  uint16_t* mc_ws  = (uint16_t*)take(SZ_MC);
  uint16_t* spt_ws = (uint16_t*)take(SZ_SPT);
  float*    zc_ws  = (float*)take(SZ_ZC);
  float*    zp_ws  = (float*)take(SZ_ZC);
  uint16_t* attn   = (uint16_t*)take(SZ_QKV);
  float*    prer   = (float*)take(SZ_PRE);
  uint16_t* omg_bf = (uint16_t*)take(SZ_OMG);
  if (off > ws_size) return;  // insufficient workspace: bail rather than corrupt

  dim3 gemm_grid(MROWS / 128, DM / 128);
  // 0) omega -> bf16 (once per call; tiny)
  cvt_omega_kernel<<<(RR * DD) / 256, 256, 0, stream>>>(omega, omg_bf);
  // 1) QKV projections (f32 in, bf16 out)
  gemm_xwt_kernel<float, uint16_t><<<gemm_grid, 256, 0, stream>>>(pre_q, wq, q_bf, MROWS, DM, DM);
  gemm_xwt_kernel<float, uint16_t><<<gemm_grid, 256, 0, stream>>>(pre_k, wk, k_bf, MROWS, DM, DM);
  gemm_xwt_kernel<float, uint16_t><<<gemm_grid, 256, 0, stream>>>(pre_v, wv, v_bf, MROWS, DM, DM);
  // 2) per-chunk k' features, k'^T V, colsums
  phase_a_kernel<<<BB * HH * NC, 256, 0, stream>>>(k_bf, v_bf, omg_bf, bvec, kp_ws, mc_ws, zc_ws);
  // 3) exclusive prefix scan over chunks -> S_prev (transposed bf16), z_prev
  scan_kernel<<<BB * HH, 256, 0, stream>>>(mc_ws, zc_ws, spt_ws, zp_ws);
  // 4) per-chunk outputs
  phase_c_kernel<<<BB * HH * NC, 256, 0, stream>>>(q_bf, v_bf, omg_bf, bvec, kp_ws, spt_ws, zp_ws, attn);
  // 5) output projection (bf16 in, f32 out)
  gemm_xwt_kernel<uint16_t, float><<<gemm_grid, 256, 0, stream>>>(attn, wo, prer, MROWS, DM, DM);
  // 6) residual + LayerNorm
  ln_kernel<<<MROWS, 256, 0, stream>>>(prer, pre_q, gamma, beta, out);
}
